// RNN_23132693856285
// MI455X (gfx1250) — matrix operand order repaired and checked by reading, not compiled, not run on hardware
//
#include <hip/hip_runtime.h>
#include <cstdint>

typedef __attribute__((ext_vector_type(16))) _Float16 v16h;
typedef __attribute__((ext_vector_type(8)))  float    v8f;
typedef __attribute__((ext_vector_type(4)))  _Float16 v4h;

#define RNN_H   1024
#define RNN_T   256
#define RNN_B   128
#define RNN_IN  512
#define RNN_OUT 512
#define NUM_E   819   // excitatory rows

// ---- workspace layout (bytes) ----
#define OFF_WXHT   ((size_t)0)                                    // [H=1024][IN=512] f16 (transposed |Wxh|)
#define OFF_WHHT   ((size_t)1048576)                              // [H][H] f16 (transposed ei*|Whh|, diag=0)
#define OFF_WOUTT  (OFF_WHHT + (size_t)2097152)                   // [OUT=512][H] f16 (transposed ei*|Wout|)
#define OFF_XH     (OFF_WOUTT + (size_t)1048576)                  // [B*T=32768][IN] f16
#define OFF_XPROJ  (OFF_XH + (size_t)32768*512*2)                 // [B*T][H] f32
#define OFF_STATES (OFF_XPROJ + (size_t)32768*1024*4)             // [B*T][H] f16

#define BCOL_STRIDE 40   // halves per staged B column (32 data + 8 pad -> 80B rows)

// ---------------------------------------------------------------------------
// WMMA lane mappings (wave32, CDNA5 ISA 7.12.2)
// A (f16 16x32): row = lane&15; kb = (lane>>4)*8; halves 0..7 = K[kb..kb+7],
//                halves 8..15 = K[kb+16..kb+23]  -> two 16B loads.
// B (f16 32x16): col = lane&15; kq = (lane>>4)*8; halves 0..7 = K[kq..kq+7], 8..15 = K[kq+16..kq+23].
// C/D (f32 16x16): col n = lane&15; VGPR r holds row m = r + 8*(lane>=16).
// ---------------------------------------------------------------------------
__device__ __forceinline__ v16h load_a16x32(const _Float16* Abase, int lda, int k, int lane) {
    const int row = lane & 15;
    const int kb  = (lane >> 4) << 3;
    const _Float16* p = Abase + (size_t)row * lda + k + kb;
    union { v16h v; float4 q[2]; } u;
    u.q[0] = *(const float4*)(p);
    u.q[1] = *(const float4*)(p + 16);
    return u.v;
}

// read a staged 32x16 B tile for column-group j from LDS (ds_load_b128 x2)
__device__ __forceinline__ v16h load_b_lds(const _Float16* lbuf, int j, int lane) {
    const int kq  = (lane >> 4) << 3;
    const int col = j * 16 + (lane & 15);
    const _Float16* p = lbuf + col * BCOL_STRIDE + kq;
    union { v16h v; float4 q[2]; } u;
    u.q[0] = *(const float4*)(p);
    u.q[1] = *(const float4*)(p + 16);
    return u.v;
}

// cooperative async stage of a 64-col x 32-K f16 B panel into LDS.
// 128 threads x 2 chunks x 16B = 4KB. ASYNCcnt-tracked (gfx1250 async path).
__device__ __forceinline__ void stage_b_async(const _Float16* __restrict__ BT, int ldb,
                                              int n0, int k, _Float16* lbuf, int tid) {
#pragma unroll
    for (int j = 0; j < 2; ++j) {
        const int c   = tid * 2 + j;      // 0..255 chunk id
        const int col = c >> 2;           // 0..63
        const int kc  = (c & 3) * 8;      // 0,8,16,24 (halves)
        const _Float16* g = BT + (size_t)(n0 + col) * ldb + k + kc;
        const uint32_t loff = (uint32_t)(uintptr_t)(lbuf + col * BCOL_STRIDE + kc);
        asm volatile("global_load_async_to_lds_b128 %0, %1, off"
                     :: "v"(loff), "v"(g) : "memory");
    }
}

__device__ __forceinline__ void wait_async_barrier() {
    asm volatile("s_wait_asynccnt 0x0" ::: "memory");
    __syncthreads();
}

// double-buffered K loop: prefetch next B panel while WMMAs consume current one.
// All 4 B tiles are pulled into distinct registers first so the backend emits
// one ds_load clause + a single s_wait_dscnt, then 4 back-to-back WMMAs.
template <int KTOT>
__device__ __forceinline__ void gemm_k_loop(const _Float16* Abase, int lda,
                                            const _Float16* __restrict__ BT, int ldb, int n0,
                                            _Float16* lds0, _Float16* lds1,
                                            int tid, int lane, v8f acc[4]) {
    stage_b_async(BT, ldb, n0, 0, lds0, tid);
    wait_async_barrier();
    _Float16* cur = lds0;
    _Float16* nxt = lds1;
    for (int k = 0; k < KTOT; k += 32) {
        if (k + 32 < KTOT) stage_b_async(BT, ldb, n0, k + 32, nxt, tid);
        v16h a  = load_a16x32(Abase, lda, k, lane);
        v16h b0 = load_b_lds(cur, 0, lane);
        v16h b1 = load_b_lds(cur, 1, lane);
        v16h b2 = load_b_lds(cur, 2, lane);
        v16h b3 = load_b_lds(cur, 3, lane);
        acc[0] = __builtin_amdgcn_wmma_f32_16x16x32_f16(false, a, false, b0,
                                                        (short)0, acc[0], false, false);
        acc[1] = __builtin_amdgcn_wmma_f32_16x16x32_f16(false, a, false, b1,
                                                        (short)0, acc[1], false, false);
        acc[2] = __builtin_amdgcn_wmma_f32_16x16x32_f16(false, a, false, b2,
                                                        (short)0, acc[2], false, false);
        acc[3] = __builtin_amdgcn_wmma_f32_16x16x32_f16(false, a, false, b3,
                                                        (short)0, acc[3], false, false);
        wait_async_barrier();
        _Float16* t = cur; cur = nxt; nxt = t;
    }
}

__device__ __forceinline__ float retanh(float x) {
    float t = tanhf(x);
    return t > 0.0f ? t : 0.0f;
}

// ---------------------------------------------------------------------------
// Kernel 1: effective weights -> |.|, EI sign, zero diagonal, transposed f16
// ---------------------------------------------------------------------------
__global__ void prep_weights_kernel(const float* __restrict__ Wxh,
                                    const float* __restrict__ Whh,
                                    const float* __restrict__ Wout,
                                    _Float16* __restrict__ WxhT,
                                    _Float16* __restrict__ WhhT,
                                    _Float16* __restrict__ WoutT) {
    const int NWXH = RNN_IN * RNN_H;     // 524288
    const int NWHH = RNN_H * RNN_H;      // 1048576
    const int NWOUT = RNN_H * RNN_OUT;   // 524288
    int idx = blockIdx.x * blockDim.x + threadIdx.x;
    if (idx < NWXH) {
        int h = idx & (RNN_H - 1);
        int i = idx >> 10;                       // Wxh[i][h]
        WxhT[(size_t)h * RNN_IN + i] = (_Float16)fabsf(Wxh[idx]);
    } else if (idx < NWXH + NWHH) {
        int j = idx - NWXH;
        int n = j & (RNN_H - 1);
        int k = j >> 10;                         // Whh[k][n]
        float v = fabsf(Whh[j]);
        float s = (k < NUM_E) ? v : -v;
        WhhT[(size_t)n * RNN_H + k] = (_Float16)((k == n) ? 0.0f : s);
    } else if (idx < NWXH + NWHH + NWOUT) {
        int j = idx - NWXH - NWHH;
        int o = j & (RNN_OUT - 1);
        int h = j >> 9;                          // Wout[h][o]
        float v = fabsf(Wout[j]);
        WoutT[(size_t)o * RNN_H + h] = (_Float16)((h < NUM_E) ? v : -v);
    }
}

// ---------------------------------------------------------------------------
// Kernel 2: x (f32) -> x_h (f16), 4 elements/thread
// ---------------------------------------------------------------------------
__global__ void convert_x_kernel(const float* __restrict__ x, _Float16* __restrict__ xh) {
    size_t idx = (size_t)blockIdx.x * blockDim.x + threadIdx.x;   // 4,194,304 threads
    float4 v = ((const float4*)x)[idx];
    v4h h;
    h.x = (_Float16)v.x; h.y = (_Float16)v.y; h.z = (_Float16)v.z; h.w = (_Float16)v.w;
    ((v4h*)xh)[idx] = h;
}

// ---------------------------------------------------------------------------
// Kernel 3: xproj = Xh[32768x512] @ Wxh_e + noise + bias   (f32 out)
// Block = 4 waves sharing one 64-col B panel, 4 consecutive M tiles.
// grid = 512 mtile-groups x 16 ngroups = 8192 blocks x 128 threads.
// ---------------------------------------------------------------------------
__global__ void __launch_bounds__(128)
xproj_gemm_kernel(const _Float16* __restrict__ Xh, const _Float16* __restrict__ WxhT,
                  const float* __restrict__ noise, const float* __restrict__ bias,
                  float* __restrict__ xproj) {
    __shared__ _Float16 lds_b[2][64 * BCOL_STRIDE];
    const int K = RNN_IN, H = RNN_H;
    const int tid  = threadIdx.x;
    const int lane = tid & 31;
    const int wave = tid >> 5;
    const int ngrp = blockIdx.x & 15;           // 16 groups of 64 cols
    const int mt   = (blockIdx.x >> 4) * 4 + wave;  // 0..2047 row tiles
    const int n0 = ngrp * 64;
    const int ncol  = lane & 15;
    const int mbase = (lane >> 4) << 3;

    v8f acc[4];
#pragma unroll
    for (int j = 0; j < 4; ++j) {
        const int n = n0 + j * 16 + ncol;
        const float bv = bias[n];
#pragma unroll
        for (int r = 0; r < 8; ++r) {
            const int m = mt * 16 + mbase + r;
            acc[j][r] = noise[(size_t)m * H + n] + bv;
        }
    }
    const _Float16* Abase = Xh + (size_t)mt * 16 * K;
    gemm_k_loop<RNN_IN>(Abase, K, WxhT, K, n0, lds_b[0], lds_b[1], tid, lane, acc);
#pragma unroll
    for (int j = 0; j < 4; ++j) {
        const int n = n0 + j * 16 + ncol;
#pragma unroll
        for (int r = 0; r < 8; ++r) {
            const int m = mt * 16 + mbase + r;
            xproj[(size_t)m * H + n] = acc[j][r];
        }
    }
}

// ---------------------------------------------------------------------------
// Kernel 4: t = 0 has h_prev = 0 -> states[:,0,:] = retanh(xproj[:,0,:])
// ---------------------------------------------------------------------------
__global__ void act0_kernel(const float* __restrict__ xproj, _Float16* __restrict__ states) {
    int idx = blockIdx.x * blockDim.x + threadIdx.x;  // 131072 = 128*1024
    int b = idx >> 10;
    int h = idx & (RNN_H - 1);
    size_t off = ((size_t)b * RNN_T) * RNN_H + h;     // t = 0
    states[off] = (_Float16)retanh(xproj[off]);
}

// ---------------------------------------------------------------------------
// Kernel 5 (x255): one recurrence step.
//   pre = h[t-1] @ Whh_e + xproj[t];  h[t] = retanh(pre)  (stored f16)
// grid = 2 mtile-groups x 16 ngroups = 32 blocks x 128 threads.
// ---------------------------------------------------------------------------
__global__ void __launch_bounds__(128)
rnn_step_kernel(_Float16* __restrict__ states, const _Float16* __restrict__ WhhT,
                const float* __restrict__ xproj, int t) {
    __shared__ _Float16 lds_b[2][64 * BCOL_STRIDE];
    const int K = RNN_H;
    const int tid  = threadIdx.x;
    const int lane = tid & 31;
    const int wave = tid >> 5;
    const int ngrp = blockIdx.x & 15;                // 16 groups of 64 cols
    const int mt   = (blockIdx.x >> 4) * 4 + wave;   // 0..7 row tiles over B=128
    const int n0 = ngrp * 64;
    const int ncol  = lane & 15;
    const int mbase = (lane >> 4) << 3;
    const int ldrow = RNN_T * RNN_H;                 // stride between batch rows

    v8f acc[4];
#pragma unroll
    for (int j = 0; j < 4; ++j) {
        const int n = n0 + j * 16 + ncol;
#pragma unroll
        for (int r = 0; r < 8; ++r) {
            const int bidx = mt * 16 + mbase + r;
            acc[j][r] = xproj[((size_t)bidx * RNN_T + t) * RNN_H + n];
        }
    }
    const _Float16* Abase = states + (size_t)mt * 16 * ldrow + (size_t)(t - 1) * RNN_H;
    gemm_k_loop<RNN_H>(Abase, ldrow, WhhT, K, n0, lds_b[0], lds_b[1], tid, lane, acc);
#pragma unroll
    for (int j = 0; j < 4; ++j) {
        const int n = n0 + j * 16 + ncol;
#pragma unroll
        for (int r = 0; r < 8; ++r) {
            const int bidx = mt * 16 + mbase + r;
            states[((size_t)bidx * RNN_T + t) * RNN_H + n] = (_Float16)retanh(acc[j][r]);
        }
    }
}

// ---------------------------------------------------------------------------
// Kernel 6: logits = states[32768x1024] @ Wout_e + bias  (f32 to d_out)
// grid = 512 mtile-groups x 8 ngroups = 4096 blocks x 128 threads.
// ---------------------------------------------------------------------------
__global__ void __launch_bounds__(128)
out_gemm_kernel(const _Float16* __restrict__ states, const _Float16* __restrict__ WoutT,
                const float* __restrict__ bias, float* __restrict__ out) {
    __shared__ _Float16 lds_b[2][64 * BCOL_STRIDE];
    const int K = RNN_H, OUT = RNN_OUT;
    const int tid  = threadIdx.x;
    const int lane = tid & 31;
    const int wave = tid >> 5;
    const int ngrp = blockIdx.x & 7;                 // 8 groups of 64 cols
    const int mt   = (blockIdx.x >> 3) * 4 + wave;   // 0..2047 row tiles
    const int n0 = ngrp * 64;
    const int ncol  = lane & 15;
    const int mbase = (lane >> 4) << 3;

    v8f acc[4];
#pragma unroll
    for (int j = 0; j < 4; ++j) {
        const float bv = bias[n0 + j * 16 + ncol];
#pragma unroll
        for (int r = 0; r < 8; ++r) acc[j][r] = bv;
    }
    const _Float16* Abase = states + (size_t)mt * 16 * K;
    gemm_k_loop<RNN_H>(Abase, K, WoutT, K, n0, lds_b[0], lds_b[1], tid, lane, acc);
#pragma unroll
    for (int j = 0; j < 4; ++j) {
        const int n = n0 + j * 16 + ncol;
#pragma unroll
        for (int r = 0; r < 8; ++r) {
            const int m = mt * 16 + mbase + r;
            out[(size_t)m * OUT + n] = acc[j][r];
        }
    }
}

// ---------------------------------------------------------------------------
extern "C" void kernel_launch(void* const* d_in, const int* in_sizes, int n_in,
                              void* d_out, int out_size, void* d_ws, size_t ws_size,
                              hipStream_t stream) {
    (void)in_sizes; (void)n_in; (void)out_size; (void)ws_size;
    const float* x         = (const float*)d_in[0];   // [128,256,512]
    const float* noise     = (const float*)d_in[1];   // [128,256,1024]
    const float* Wxh       = (const float*)d_in[2];   // [512,1024]
    const float* Whh       = (const float*)d_in[3];   // [1024,1024]
    const float* Wout      = (const float*)d_in[4];   // [1024,512]
    const float* Wh_bias   = (const float*)d_in[5];   // [1,1024]
    const float* Wout_bias = (const float*)d_in[6];   // [1,512]
    float* out = (float*)d_out;

    char* ws = (char*)d_ws;
    _Float16* WxhT   = (_Float16*)(ws + OFF_WXHT);
    _Float16* WhhT   = (_Float16*)(ws + OFF_WHHT);
    _Float16* WoutT  = (_Float16*)(ws + OFF_WOUTT);
    _Float16* Xh     = (_Float16*)(ws + OFF_XH);
    float*    xproj  = (float*)   (ws + OFF_XPROJ);
    _Float16* states = (_Float16*)(ws + OFF_STATES);

    // 1) effective weights (2,097,152 elements total)
    prep_weights_kernel<<<8192, 256, 0, stream>>>(Wxh, Whh, Wout, WxhT, WhhT, WoutT);
    // 2) x -> f16 (16,777,216 elems, 4/thread)
    convert_x_kernel<<<16384, 256, 0, stream>>>(x, Xh);
    // 3) input projection GEMM (+noise +bias)
    xproj_gemm_kernel<<<8192, 128, 0, stream>>>(Xh, WxhT, noise, Wh_bias, xproj);
    // 4) t = 0 (h0 = 0)
    act0_kernel<<<512, 256, 0, stream>>>(xproj, states);
    // 5) recurrence, 255 dependent steps
    for (int t = 1; t < RNN_T; ++t)
        rnn_step_kernel<<<32, 128, 0, stream>>>(states, WhhT, xproj, t);
    // 6) output projection GEMM
    out_gemm_kernel<<<4096, 128, 0, stream>>>(states, WoutT, Wout_bias, out);
}
